// IA_layer3_38568806318691
// MI455X (gfx1250) — compile-verified, matrix operand order checked by reading, not run on hardware
//
#include <hip/hip_runtime.h>

typedef __bf16 bf16_t;
typedef __attribute__((ext_vector_type(16))) __bf16 v16bf;
typedef __attribute__((ext_vector_type(8)))  float  v8f;

constexpr int   kB = 8;
constexpr int   kC = 256;   // IC == PC == 256
constexpr int   kN = 2048;
constexpr float kScale = 0.0625f;  // 1/sqrt(256)

// ---------------------------------------------------------------------------
// Tile loader: 16x32 bf16 tile in WMMA A/B layout from a row-major matrix
// (rows = M (A) or N (B), cols = K, leading dim `ld` in elements).
// Lane l: row = l&15, k-half = l>>4. Two 16-byte chunks per lane:
//   chunk0 = k in [8*kh, 8*kh+8), chunk1 = k in [16+8*kh, 16+8*kh+8)
// which matches the ISA 16-bit A-matrix layout (VGPR0..3 / VGPR4..7).
// ---------------------------------------------------------------------------
__device__ inline v16bf load_tile(const bf16_t* __restrict__ base, int ld) {
  const int lane = threadIdx.x & 31;
  const int m    = lane & 15;
  const int kh   = lane >> 4;
  const bf16_t* p = base + (size_t)m * ld + kh * 8;
  union { uint4 q[2]; v16bf v; } u;
  u.q[0] = *(const uint4*)(p);
  u.q[1] = *(const uint4*)(p + 16);
  return u.v;
}

__device__ inline v8f wmma_bf16(v16bf a, v16bf b, v8f c) {
  return __builtin_amdgcn_wmma_f32_16x16x32_bf16(
      /*neg_a=*/false, a, /*neg_b=*/false, b,
      /*c_mod=*/(short)0, c, /*reuse_a=*/false, /*reuse_b=*/false);
}

__device__ inline void atomicMaxF(float* addr, float val) {
  int* ai  = (int*)addr;
  int  old = __float_as_int(*addr);
  while (__int_as_float(old) < val) {
    int prev = atomicCAS(ai, old, __float_as_int(val));
    if (prev == old) break;
    old = prev;
  }
}

// ---------------------------------------------------------------------------
// Kernel 0: bf16 conversions / transposes + stats init.
//   rp  [B][N][C] = point^T ; rpT [B][C][N] = point ; imgT[B][N][C] = img^T
//   wb  [C][C]    = w       ; stats (rowmax/colmax/-inf, rowsum/colsum/0)
// ---------------------------------------------------------------------------
__global__ __launch_bounds__(256) void convert_kernel(
    const float* __restrict__ img, const float* __restrict__ pt,
    const float* __restrict__ w,
    bf16_t* __restrict__ rp,  bf16_t* __restrict__ rpT,
    bf16_t* __restrict__ imgT, bf16_t* __restrict__ wb,
    float* __restrict__ rowmax, float* __restrict__ rowsum,
    float* __restrict__ colmax, float* __restrict__ colsum) {
  const int gid = blockIdx.x * 256 + threadIdx.x;
  const int total = kB * kC * kN;
  if (gid < total) {
    const int n = gid % kN;
    const int c = (gid / kN) % kC;
    const int b = gid / (kN * kC);
    const float pv = pt[gid];
    const float iv = img[gid];
    rpT[gid] = (bf16_t)pv;
    rp  [((size_t)b * kN + n) * kC + c] = (bf16_t)pv;
    imgT[((size_t)b * kN + n) * kC + c] = (bf16_t)iv;
  }
  if (gid < kC * kC) wb[gid] = (bf16_t)w[gid];
  if (gid < kB * kN) {
    rowmax[gid] = -3.0e38f; colmax[gid] = -3.0e38f;
    rowsum[gid] = 0.0f;     colsum[gid] = 0.0f;
  }
}

// ---------------------------------------------------------------------------
// Kernel 1: ri[b][n][d] = sum_c imgT[b][n][c] * w[d][c] + bias[d]
// One wave -> 32x32 tile (2x2 WMMA, K=256). Writes ri and riT (bf16).
// ---------------------------------------------------------------------------
__global__ __launch_bounds__(256) void ri_gemm_kernel(
    const bf16_t* __restrict__ imgT, const bf16_t* __restrict__ wb,
    const float* __restrict__ bias,
    bf16_t* __restrict__ ri, bf16_t* __restrict__ riT) {
  const int wv   = (blockIdx.x * blockDim.x + threadIdx.x) >> 5;
  const int lane = threadIdx.x & 31;
  const int dt = wv % (kC / 32);
  const int nt = (wv / (kC / 32)) % (kN / 32);
  const int b  = wv / ((kC / 32) * (kN / 32));
  const int n0 = nt * 32, d0 = dt * 32;

  const bf16_t* Ab = imgT + ((size_t)b * kN + n0) * kC;
  const bf16_t* Bb = wb   + (size_t)d0 * kC;

  v8f acc[2][2] = {};
  for (int k = 0; k < kC; k += 32) {
    v16bf a0 = load_tile(Ab + k, kC);
    v16bf a1 = load_tile(Ab + (size_t)16 * kC + k, kC);
    v16bf b0 = load_tile(Bb + k, kC);
    v16bf b1 = load_tile(Bb + (size_t)16 * kC + k, kC);
    acc[0][0] = wmma_bf16(a0, b0, acc[0][0]);
    acc[0][1] = wmma_bf16(a0, b1, acc[0][1]);
    acc[1][0] = wmma_bf16(a1, b0, acc[1][0]);
    acc[1][1] = wmma_bf16(a1, b1, acc[1][1]);
  }

  const int j = lane & 15, ih = lane >> 4;
  for (int ti = 0; ti < 2; ++ti)
    for (int tj = 0; tj < 2; ++tj) {
      const int dg = d0 + tj * 16 + j;
      const float bv = bias[dg];
      for (int r = 0; r < 8; ++r) {
        const int ng = n0 + ti * 16 + ih * 8 + r;
        const float v = acc[ti][tj][r] + bv;
        ri [((size_t)b * kN + ng) * kC + dg] = (bf16_t)v;
        riT[((size_t)b * kC + dg) * kN + ng] = (bf16_t)v;
      }
    }
}

// ---------------------------------------------------------------------------
// Helper: one wave computes the scaled 32x32 S tile
//   S[i][j] = scale * sum_c A[n0+i][c] * Bm[m0+j][c]
// ---------------------------------------------------------------------------
__device__ inline void s_tile(const bf16_t* __restrict__ Ab,
                              const bf16_t* __restrict__ Bb,
                              v8f s[2][2]) {
  s[0][0] = v8f{}; s[0][1] = v8f{}; s[1][0] = v8f{}; s[1][1] = v8f{};
  for (int k = 0; k < kC; k += 32) {
    v16bf a0 = load_tile(Ab + k, kC);
    v16bf a1 = load_tile(Ab + (size_t)16 * kC + k, kC);
    v16bf b0 = load_tile(Bb + k, kC);
    v16bf b1 = load_tile(Bb + (size_t)16 * kC + k, kC);
    s[0][0] = wmma_bf16(a0, b0, s[0][0]);
    s[0][1] = wmma_bf16(a0, b1, s[0][1]);
    s[1][0] = wmma_bf16(a1, b0, s[1][0]);
    s[1][1] = wmma_bf16(a1, b1, s[1][1]);
  }
  s[0][0] *= kScale; s[0][1] *= kScale; s[1][0] *= kScale; s[1][1] *= kScale;
}

// ---------------------------------------------------------------------------
// Kernel 2: row (n) and col (m) max of S via atomics. One wave per 32x32 tile.
// ---------------------------------------------------------------------------
__global__ __launch_bounds__(256) void stats_max_kernel(
    const bf16_t* __restrict__ rp, const bf16_t* __restrict__ ri,
    float* __restrict__ rmax, float* __restrict__ cmax) {
  const int wv   = (blockIdx.x * blockDim.x + threadIdx.x) >> 5;
  const int lane = threadIdx.x & 31;
  const int mt = wv % (kN / 32);
  const int nt = (wv / (kN / 32)) % (kN / 32);
  const int b  = wv / ((kN / 32) * (kN / 32));
  const int n0 = nt * 32, m0 = mt * 32;
  const int j = lane & 15, ih = lane >> 4;

  v8f s[2][2];
  s_tile(rp + ((size_t)b * kN + n0) * kC, ri + ((size_t)b * kN + m0) * kC, s);

  // column (m-index) max: reduce over i = vgprs + xor16
  for (int tj = 0; tj < 2; ++tj) {
    float cm = -3.0e38f;
    for (int ti = 0; ti < 2; ++ti)
      for (int r = 0; r < 8; ++r) cm = fmaxf(cm, s[ti][tj][r]);
    cm = fmaxf(cm, __shfl_xor(cm, 16, 32));
    if (ih == 0) atomicMaxF(&cmax[(size_t)b * kN + m0 + tj * 16 + j], cm);
  }
  // row (n-index) max: reduce over j = lane butterfly within 16-lane halves
  for (int ti = 0; ti < 2; ++ti)
    for (int r = 0; r < 8; ++r) {
      float rm = fmaxf(s[ti][0][r], s[ti][1][r]);
      rm = fmaxf(rm, __shfl_xor(rm, 1, 32));
      rm = fmaxf(rm, __shfl_xor(rm, 2, 32));
      rm = fmaxf(rm, __shfl_xor(rm, 4, 32));
      rm = fmaxf(rm, __shfl_xor(rm, 8, 32));
      if (j == 0)
        atomicMaxF(&rmax[(size_t)b * kN + n0 + ti * 16 + ih * 8 + r], rm);
    }
}

// ---------------------------------------------------------------------------
// Kernel 3: rowsum[n] = sum_m exp(S-rmax[n]); colsum[m] = sum_n exp(S-cmax[m])
// ---------------------------------------------------------------------------
__global__ __launch_bounds__(256) void stats_sum_kernel(
    const bf16_t* __restrict__ rp, const bf16_t* __restrict__ ri,
    const float* __restrict__ rmax, const float* __restrict__ cmax,
    float* __restrict__ rsum, float* __restrict__ csum) {
  const int wv   = (blockIdx.x * blockDim.x + threadIdx.x) >> 5;
  const int lane = threadIdx.x & 31;
  const int mt = wv % (kN / 32);
  const int nt = (wv / (kN / 32)) % (kN / 32);
  const int b  = wv / ((kN / 32) * (kN / 32));
  const int n0 = nt * 32, m0 = mt * 32;
  const int j = lane & 15, ih = lane >> 4;

  v8f s[2][2];
  s_tile(rp + ((size_t)b * kN + n0) * kC, ri + ((size_t)b * kN + m0) * kC, s);

  float cmv[2], rmv[2][8];
  for (int tj = 0; tj < 2; ++tj)
    cmv[tj] = cmax[(size_t)b * kN + m0 + tj * 16 + j];
  for (int ti = 0; ti < 2; ++ti)
    for (int r = 0; r < 8; ++r)
      rmv[ti][r] = rmax[(size_t)b * kN + n0 + ti * 16 + ih * 8 + r];

  float cacc[2] = {0.0f, 0.0f};
  float racc[2][8] = {};
  for (int ti = 0; ti < 2; ++ti)
    for (int tj = 0; tj < 2; ++tj)
      for (int r = 0; r < 8; ++r) {
        const float sv = s[ti][tj][r];
        cacc[tj]     += __expf(sv - cmv[tj]);
        racc[ti][r]  += __expf(sv - rmv[ti][r]);
      }
  for (int tj = 0; tj < 2; ++tj) {
    float cs = cacc[tj] + __shfl_xor(cacc[tj], 16, 32);
    if (ih == 0) atomicAdd(&csum[(size_t)b * kN + m0 + tj * 16 + j], cs);
  }
  for (int ti = 0; ti < 2; ++ti)
    for (int r = 0; r < 8; ++r) {
      float rs = racc[ti][r];
      rs += __shfl_xor(rs, 1, 32);
      rs += __shfl_xor(rs, 2, 32);
      rs += __shfl_xor(rs, 4, 32);
      rs += __shfl_xor(rs, 8, 32);
      if (j == 0)
        atomicAdd(&rsum[(size_t)b * kN + n0 + ti * 16 + ih * 8 + r], rs);
    }
}

// ---------------------------------------------------------------------------
// Kernel 4 (shared by out1/out2): fused  out = softmaxed(S) @ V
//   out1: A=rp (rows n), Bm=ri (rows m), stats per m (col), VT=rpT -> out1
//   out2: A=ri (rows m), Bm=rp (rows n), stats per n (row), VT=riT -> out2
// Block = 8 waves, owns 32 output rows; loops over groups in 256-wide chunks;
// each wave builds one 32x32 prob tile (WMMA + exp), stages bf16 tile in LDS,
// then all waves GEMM the shared 32x256 prob block against their 32-wide
// d-slice of VT.
// ---------------------------------------------------------------------------
__global__ __launch_bounds__(256) void attn_out_kernel(
    const bf16_t* __restrict__ A,  const bf16_t* __restrict__ Bm,
    const bf16_t* __restrict__ VT,
    const float* __restrict__ gmax, const float* __restrict__ gsum,
    float* __restrict__ out) {
  const int wv   = threadIdx.x >> 5;   // 0..7 -> d-slice
  const int lane = threadIdx.x & 31;
  const int b  = blockIdx.y;
  const int n0 = blockIdx.x * 32;
  const int d0 = wv * 32;
  const int j = lane & 15, ih = lane >> 4;

  __shared__ bf16_t Qs[32][264];       // 32 x 256 (+8 pad), bf16

  v8f o[2][2] = {};
  const bf16_t* Ab = A + ((size_t)b * kN + n0) * kC;

  for (int mc = 0; mc < kN; mc += 256) {
    const int mg0 = mc + wv * 32;
    // ---- phase 1: this wave's 32x32 prob tile -------------------------
    v8f s[2][2];
    s_tile(Ab, Bm + ((size_t)b * kN + mg0) * kC, s);
    __syncthreads();                   // previous chunk fully consumed
    for (int tj = 0; tj < 2; ++tj) {
      const int mg = mg0 + tj * 16 + j;
      const float mx  = gmax[(size_t)b * kN + mg];
      const float inv = 1.0f / gsum[(size_t)b * kN + mg];
      for (int ti = 0; ti < 2; ++ti)
        for (int r = 0; r < 8; ++r) {
          const float q = __expf(s[ti][tj][r] - mx) * inv;
          Qs[ti * 16 + ih * 8 + r][wv * 32 + tj * 16 + j] = (bf16_t)q;
        }
    }
    __syncthreads();
    // ---- phase 2: o += Q(32x256) x VT(d-slice)^T ----------------------
    const bf16_t* Vb = VT + ((size_t)b * kC + d0) * kN + mc;
    for (int k = 0; k < 256; k += 32) {
      v16bf a0 = load_tile(&Qs[0][k], 264);
      v16bf a1 = load_tile(&Qs[16][k], 264);
      v16bf b0 = load_tile(Vb + k, kN);
      v16bf b1 = load_tile(Vb + (size_t)16 * kN + k, kN);
      o[0][0] = wmma_bf16(a0, b0, o[0][0]);
      o[0][1] = wmma_bf16(a0, b1, o[0][1]);
      o[1][0] = wmma_bf16(a1, b0, o[1][0]);
      o[1][1] = wmma_bf16(a1, b1, o[1][1]);
    }
  }

  for (int ti = 0; ti < 2; ++ti)
    for (int tj = 0; tj < 2; ++tj)
      for (int r = 0; r < 8; ++r)
        out[((size_t)b * kN + n0 + ti * 16 + ih * 8 + r) * kC + d0 + tj * 16 + j] =
            o[ti][tj][r];
}

// ---------------------------------------------------------------------------
extern "C" void kernel_launch(void* const* d_in, const int* in_sizes, int n_in,
                              void* d_out, int out_size, void* d_ws, size_t ws_size,
                              hipStream_t stream) {
  const float* img  = (const float*)d_in[0];  // [B, IC, N]
  const float* pt   = (const float*)d_in[1];  // [B, PC, N]
  const float* w    = (const float*)d_in[2];  // [PC, IC]
  const float* bias = (const float*)d_in[3];  // [PC]

  float* out1 = (float*)d_out;
  float* out2 = out1 + (size_t)kB * kN * kC;

  // workspace carve (256B-aligned)
  char* ws = (char*)d_ws;
  size_t off = 0;
  auto carve = [&](size_t bytes) {
    void* p = ws + off;
    off += (bytes + 255) & ~(size_t)255;
    return p;
  };
  const size_t EL = (size_t)kB * kN * kC;
  bf16_t* rp     = (bf16_t*)carve(EL * 2);
  bf16_t* rpT    = (bf16_t*)carve(EL * 2);
  bf16_t* imgT   = (bf16_t*)carve(EL * 2);
  bf16_t* ri     = (bf16_t*)carve(EL * 2);
  bf16_t* riT    = (bf16_t*)carve(EL * 2);
  bf16_t* wb     = (bf16_t*)carve((size_t)kC * kC * 2);
  float*  rowmax = (float*)carve((size_t)kB * kN * 4);
  float*  rowsum = (float*)carve((size_t)kB * kN * 4);
  float*  colmax = (float*)carve((size_t)kB * kN * 4);
  float*  colsum = (float*)carve((size_t)kB * kN * 4);

  // 0) conversions + stats init
  convert_kernel<<<(kB * kC * kN + 255) / 256, 256, 0, stream>>>(
      img, pt, w, rp, rpT, imgT, wb, rowmax, rowsum, colmax, colsum);

  // 1) ri GEMM: 8 * 64 * 8 = 4096 waves -> 512 blocks of 8 waves
  ri_gemm_kernel<<<(kB * (kN / 32) * (kC / 32)) / 8, 256, 0, stream>>>(
      imgT, wb, bias, ri, riT);

  // 2/3) softmax stats: 8 * 64 * 64 = 32768 waves -> 4096 blocks
  const int statsBlocks = (kB * (kN / 32) * (kN / 32)) / 8;
  stats_max_kernel<<<statsBlocks, 256, 0, stream>>>(rp, ri, rowmax, colmax);
  stats_sum_kernel<<<statsBlocks, 256, 0, stream>>>(rp, ri, rowmax, colmax,
                                                    rowsum, colsum);

  // 4) fused attention outputs
  dim3 grid(kN / 32, kB);
  // out1[n][d] = sum_m exp(S[n][m]-colmax[m])/colsum[m] * rp[m][d]
  attn_out_kernel<<<grid, 256, 0, stream>>>(rp, ri, rpT, colmax, colsum, out1);
  // out2[m][d] = sum_n exp(S[n][m]-rowmax[n])/rowsum[n] * ri[n][d]
  attn_out_kernel<<<grid, 256, 0, stream>>>(ri, rp, riT, rowmax, rowsum, out2);
}
